// VARTokenizer_77266461655427
// MI455X (gfx1250) — matrix operand order repaired and checked by reading, not compiled, not run on hardware
//
#include <hip/hip_runtime.h>
#include <hip/hip_bf16.h>

typedef _Float16 f16;
typedef __attribute__((ext_vector_type(16))) _Float16 v16h;
typedef __attribute__((ext_vector_type(8)))  float    v8f;

#define BATCH      2048
#define TOK_STRIDE 3726

__device__ __forceinline__ v8f wmma16(v16h a, v16h b, v8f c) {
  // D = A(16x32 f16) * B(32x16 f16) + C(16x16 f32)
  return __builtin_amdgcn_wmma_f32_16x16x32_f16(false, a, false, b, (short)0, c,
                                                false, false);
}

__device__ __forceinline__ float gelu_exact(float x) {
  return 0.5f * x * (1.0f + erff(x * 0.70710678118654752f));
}

// conv1 im2col K -> offset inside one batch row of the r band tile
// r layout: [m][c][RR rows][P cols];  off = (c*RR + dy)*P + dx
template <int P, int RR>
__device__ constexpr int im2off(int K) {
  if (K >= 54) return 0;                       // masked slots; any valid addr
  const int c = K / 9, rem = K % 9, dy = rem / 3, dx = rem % 3;
  return (c * RR + dy) * P + dx;
}

// ---------------------------------------------------------------------------
// Kernel 1: bilinear down-resize (half-pixel, no antialias) + FSQ -> tokens
// ---------------------------------------------------------------------------
__global__ void k_resize_fsq(const float* __restrict__ f, float* __restrict__ out,
                             int h, int off) {
  int idx = blockIdx.x * blockDim.x + threadIdx.x;
  int hw = h * h;
  int total = BATCH * 6 * hw;
  if (idx >= total) return;
  int x = idx % h;
  int y = (idx / h) % h;
  int c = (idx / hw) % 6;
  int b = idx / (6 * hw);

  float s  = 16.0f / (float)h;
  float sy = ((float)y + 0.5f) * s - 0.5f;
  float sx = ((float)x + 0.5f) * s - 0.5f;
  int   y0 = (int)floorf(sy); float ty = sy - (float)y0;
  int   x0 = (int)floorf(sx); float tx = sx - (float)x0;
  int y0c = min(15, max(0, y0)), y1c = min(15, max(0, y0 + 1));
  int x0c = min(15, max(0, x0)), x1c = min(15, max(0, x0 + 1));
  const float* fb = f + ((size_t)b * 6 + c) * 256;
  float v = (1.f - ty) * ((1.f - tx) * fb[y0c * 16 + x0c] + tx * fb[y0c * 16 + x1c])
          +        ty  * ((1.f - tx) * fb[y1c * 16 + x0c] + tx * fb[y1c * 16 + x1c]);

  // FSQ (levels [8,8,8,5,5,5], eps = 1e-3)
  float half_l, offs, inv_hw;
  if (c < 3) { half_l = 3.5035f; offs = 0.5f; inv_hw = 0.25f; }
  else       { half_l = 2.002f;  offs = 0.0f; inv_hw = 0.5f;  }
  float shift   = (offs != 0.0f) ? atanhf(offs / half_l) : 0.0f;
  float bounded = tanhf(v + shift) * half_l - offs;
  float q       = rintf(bounded);  // RNE == jnp.round
  out[(size_t)b * TOK_STRIDE + off + (c * h + y) * h + x] = q * inv_hw;
}

// ---------------------------------------------------------------------------
// Kernel 2: residual block conv3x3(6->48) -> GELU -> conv3x3(48->6), via WMMA.
// WG = 16 batch rows x one band of Hb output rows. Two phases:
//  P1: hidden rows y0-1..y0+Hb computed ONCE each (6 WMMA + GELU / position)
//      into halo-padded LDS plane hplane[16][Hb+2][W+2][48] (f16).
//  P2: conv2 per output pixel: 9 neighbors x 2 K-chunks, contiguous channel
//      runs, zero halo => branch-free, mask-free inner loop (18 WMMA / pixel).
// ---------------------------------------------------------------------------
template <int H, int Hb>
__global__ __launch_bounds__(256)
void k_resblock(const float* __restrict__ tok, int off,
                const float* __restrict__ w1, const float* __restrict__ b1,
                const float* __restrict__ w2, const float* __restrict__ b2,
                float* __restrict__ z) {
  constexpr int W = H, HW = H * W, P = W + 2;
  constexpr int HR = Hb + 2;                  // hidden rows in band
  constexpr int RR = Hb + 4;                  // r rows in band
  constexpr int NHP = 16 * HR * P * 48;       // hplane halves
  constexpr int NR  = 16 * 6 * RR * P;        // r halves

  extern __shared__ char smem[];
  f16*   hplane = (f16*)smem;                 // [m][hr][xp][ch]
  f16*   r_lds  = hplane + NHP;               // [m][c][rl][xp]
  f16*   w1lds  = r_lds + NR;                 // [n][K64]
  f16*   w2lds  = w1lds + 48 * 64;            // [nb][ch16][K64], fully zero-padded
  float* b1lds  = (float*)(w2lds + 9 * 16 * 64);
  float* b2lds  = b1lds + 48;

  int tid = threadIdx.x;
  int b0  = blockIdx.x * 16;
  int y0  = blockIdx.y * Hb;

  // ---- init: zero hidden plane (halo + OOB rows stay zero)
  uint4* hz = (uint4*)hplane;
  for (int i = tid; i < NHP / 8; i += 256) hz[i] = make_uint4(0, 0, 0, 0);

  for (int idx = tid; idx < 48 * 64; idx += 256) {
    int n = idx >> 6, K = idx & 63;
    f16 v = (f16)0.f;
    if (K < 54) {
      int c = K / 9, rem = K % 9, dy = rem / 3, dx = rem % 3;
      v = (f16)w1[((n * 6 + c) * 3 + dy) * 3 + dx];
    }
    w1lds[idx] = v;
  }
  for (int idx = tid; idx < 9 * 16 * 64; idx += 256) {
    int nb = idx / 1024, rem = idx % 1024, c = rem / 64, K = rem % 64;
    f16 v = (f16)0.f;
    if (c < 6 && K < 48) {
      int dy = nb / 3, dx = nb % 3;
      v = (f16)w2[((c * 48 + K) * 3 + dy) * 3 + dx];
    }
    w2lds[idx] = v;
  }
  if (tid < 48) b1lds[tid] = b1[tid];
  if (tid < 6)  b2lds[tid] = b2[tid];
  // r band (rows y0-2 .. y0+Hb+1), halo-padded, zero outside image
  for (int idx = tid; idx < NR; idx += 256) {
    int m = idx / (6 * RR * P), rem = idx % (6 * RR * P);
    int c = rem / (RR * P), rem2 = rem % (RR * P);
    int rl = rem2 / P, xp = rem2 % P;
    int ry = y0 - 2 + rl, rx = xp - 1;
    f16 v = (f16)0.f;
    if (ry >= 0 && ry < H && rx >= 0 && rx < W)
      v = (f16)tok[(size_t)(b0 + m) * TOK_STRIDE + off + (c * H + ry) * W + rx];
    r_lds[idx] = v;
  }
  __syncthreads();

  int wave = tid >> 5, lane = tid & 31;
  int g = (lane >> 4) & 1, col = lane & 15;
  int mbase_r = col * 6 * RR * P;             // A row base (M = col)

  // per-lane conv1 im2col offsets: compile-time constants selected by g
  int off1[32];
  #pragma unroll
  for (int s = 0; s < 32; ++s) {
    const int kc = (s / 16) * 32, t = s % 16;
    const int K0 = kc + t + (t >= 8 ? 8 : 0);
    off1[s] = g ? im2off<P, RR>(K0 + 8) : im2off<P, RR>(K0);
  }

  int swave = __builtin_amdgcn_readfirstlane((int)tid) >> 5;

  // ============ Phase 1: hidden rows (each position computed once) ==========
  for (int p = swave; p < HR * W; p += 8) {
    int hr = p / W, hx = p % W;
    int hy = y0 - 1 + hr;
    if (hy < 0 || hy >= H) continue;          // scalar; rows stay zero

    int pix = hr * P + hx;                    // rl = hr+dy, xp = hx+dx via off1
    v8f h0, h1, h2;
    #pragma unroll
    for (int j = 0; j < 8; ++j) { h0[j] = 0.f; h1[j] = 0.f; h2[j] = 0.f; }

    #pragma unroll
    for (int kc = 0; kc < 64; kc += 32) {
      const int sb = (kc / 32) * 16;
      v16h a;
      #pragma unroll
      for (int t = 0; t < 16; ++t) {
        f16 v = r_lds[mbase_r + pix + off1[sb + t]];   // unconditional ds load
        if (kc == 32) {                                // compile-time slot masks
          if (t >= 14)     v = (f16)0.f;               // K = 54,55 / 62,63
          else if (t >= 8) v = g ? (f16)0.f : v;       // g=1: K = 56..61
        }
        a[t] = v;
      }
      const f16* p0 = &w1lds[(0 * 16 + col) * 64 + kc + 8 * g];
      const f16* p1 = &w1lds[(1 * 16 + col) * 64 + kc + 8 * g];
      const f16* p2 = &w1lds[(2 * 16 + col) * 64 + kc + 8 * g];
      v16h bA, bB, bC;
      #pragma unroll
      for (int t = 0; t < 16; ++t) { const int q = t + (t >= 8 ? 8 : 0); bA[t] = p0[q]; bB[t] = p1[q]; bC[t] = p2[q]; }
      h0 = wmma16(a, bA, h0);
      h1 = wmma16(a, bB, h1);
      h2 = wmma16(a, bC, h2);
    }

    // bias + exact GELU -> hplane (once per hidden value)
    #pragma unroll
    for (int j = 0; j < 8; ++j) {
      int M = j + 8 * g;                      // C layout: lanes<16 M=j else M=j+8
      int hbase = ((M * HR + hr) * P + (hx + 1)) * 48;
      hplane[hbase +  0 + col] = (f16)gelu_exact(h0[j] + b1lds[ 0 + col]);
      hplane[hbase + 16 + col] = (f16)gelu_exact(h1[j] + b1lds[16 + col]);
      hplane[hbase + 32 + col] = (f16)gelu_exact(h2[j] + b1lds[32 + col]);
    }
  }
  __syncthreads();

  // ============ Phase 2: conv2, branch-free (halo provides zero padding) ====
  for (int p = swave; p < Hb * W; p += 8) {
    int yl = p / W, x = p % W;
    v8f acc;
    #pragma unroll
    for (int j = 0; j < 8; ++j) acc[j] = 0.f;

    #pragma unroll
    for (int nb = 0; nb < 9; ++nb) {
      const int dy = nb / 3, dx = nb % 3;
      const f16* ap = &hplane[((col * HR + yl + dy) * P + x + dx) * 48 + 8 * g];
      const f16* b0p = &w2lds[(nb * 16 + col) * 64 + 8 * g];

      #pragma unroll
      for (int kc = 0; kc < 64; kc += 32) {
        v16h a2;
        #pragma unroll
        for (int t = 0; t < 8; ++t)  a2[t] = ap[kc + t];          // K=kc+8g+t
        #pragma unroll
        for (int t = 8; t < 16; ++t)
          a2[t] = (kc == 32) ? (f16)0.f : ap[kc + t + 8];         // K>=48 -> 0
        v16h bv;
        #pragma unroll
        for (int t = 0; t < 16; ++t) bv[t] = b0p[kc + t + (t >= 8 ? 8 : 0)];         // zero-padded
        acc = wmma16(a2, bv, acc);
      }
    }

    // z = conv2 + b2  (layout [B][6][H][W])
    if (col < 6) {
      #pragma unroll
      for (int j = 0; j < 8; ++j) {
        int M = j + 8 * g;
        z[((size_t)(b0 + M) * 6 + col) * HW + (y0 + yl) * W + x] = acc[j] + b2lds[col];
      }
    }
  }
}

template <int H, int Hb>
constexpr size_t rb_smem() {
  constexpr int P = H + 2, HR = Hb + 2, RR = Hb + 4;
  return ((size_t)16 * HR * P * 48 + 16 * 6 * RR * P + 48 * 64 + 9 * 16 * 64) * 2
         + (48 + 6) * 4 + 16;
}

// ---------------------------------------------------------------------------
// Kernel 3: f -= bilinear_upsample(z, (h,w)->(16,16))   (gather form)
// ---------------------------------------------------------------------------
__global__ void k_upsub(float* __restrict__ f, const float* __restrict__ z, int h) {
  int idx = blockIdx.x * blockDim.x + threadIdx.x;
  if (idx >= BATCH * 6 * 256) return;
  int X = idx & 15, Y = (idx >> 4) & 15, c = (idx >> 8) % 6, b = idx / (6 * 256);
  float s  = (float)h / 16.0f;
  float sy = ((float)Y + 0.5f) * s - 0.5f;
  float sx = ((float)X + 0.5f) * s - 0.5f;
  int   y0 = (int)floorf(sy); float ty = sy - (float)y0;
  int   x0 = (int)floorf(sx); float tx = sx - (float)x0;
  int y0c = min(h - 1, max(0, y0)), y1c = min(h - 1, max(0, y0 + 1));
  int x0c = min(h - 1, max(0, x0)), x1c = min(h - 1, max(0, x0 + 1));
  const float* zb = z + ((size_t)b * 6 + c) * h * h;
  float v = (1.f - ty) * ((1.f - tx) * zb[y0c * h + x0c] + tx * zb[y0c * h + x1c])
          +        ty  * ((1.f - tx) * zb[y1c * h + x0c] + tx * zb[y1c * h + x1c]);
  f[idx] -= v;
}

// ---------------------------------------------------------------------------
extern "C" void kernel_launch(void* const* d_in, const int* in_sizes, int n_in,
                              void* d_out, int out_size, void* d_ws, size_t ws_size,
                              hipStream_t stream) {
  const float* latents = (const float*)d_in[0];
  const float* w1 = (const float*)d_in[1];
  const float* b1 = (const float*)d_in[2];
  const float* w2 = (const float*)d_in[3];
  const float* b2 = (const float*)d_in[4];
  float* out = (float*)d_out;
  float* f = (float*)d_ws;                         // [B][6][16][16] residual latent
  float* z = f + (size_t)BATCH * 6 * 256;          // [B][6][h][w] scratch (max 12x12)

  hipMemcpyAsync(f, latents, sizeof(float) * BATCH * 6 * 256,
                 hipMemcpyDeviceToDevice, stream);

  static const int HS[8]  = {1, 2, 4, 6, 8, 10, 12, 16};
  static const int OFF[8] = {0, 6, 30, 126, 342, 726, 1326, 2190};

  for (int i = 0; i < 8; ++i) {
    int h = HS[i];
    int total = BATCH * 6 * h * h;
    k_resize_fsq<<<(total + 255) / 256, 256, 0, stream>>>(f, out, h, OFF[i]);
    if (i < 7) {
      const float* w1i = w1 + (size_t)i * 48 * 6 * 9;
      const float* b1i = b1 + (size_t)i * 48;
      const float* w2i = w2 + (size_t)i * 6 * 48 * 9;
      const float* b2i = b2 + (size_t)i * 6;

      #define LAUNCH_RB(HH, HBB)                                                  \
        k_resblock<HH, HBB><<<dim3(BATCH / 16, HH / HBB), 256,                    \
                              rb_smem<HH, HBB>(), stream>>>(                      \
            out, OFF[i], w1i, b1i, w2i, b2i, z)
      switch (h) {
        case 1:  LAUNCH_RB(1, 1);  break;   //  ~41 KB LDS
        case 2:  LAUNCH_RB(2, 2);  break;   //  ~54 KB
        case 4:  LAUNCH_RB(4, 4);  break;   //  ~89 KB
        case 6:  LAUNCH_RB(6, 6);  break;   // ~138 KB
        case 8:  LAUNCH_RB(8, 4);  break;   // ~132 KB, 2 bands
        case 10: LAUNCH_RB(10, 5); break;   // ~175 KB, 2 bands
        case 12: LAUNCH_RB(12, 3); break;   // ~151 KB, 4 bands
      }
      #undef LAUNCH_RB
      k_upsub<<<(BATCH * 6 * 256 + 255) / 256, 256, 0, stream>>>(f, z, h);
    }
  }
}
